// CocrystalPredictor_32315333935106
// MI455X (gfx1250) — matrix operand order repaired and checked by reading, not compiled, not run on hardware
//
#include <hip/hip_runtime.h>
#include <hip/hip_bf16.h>
#include <stdint.h>

// ---------------- problem constants (match reference) ----------------
#define N_NODES 150000
#define N_EDGES 600000
#define N_TOT   (N_EDGES + N_NODES)   // real edges + self loops
#define NBATCH  3072
#define NODE_DIM 64
#define HID 128
#define EMBD 36
#define GDIM 200
#define BN_SCALE 0.99999500003750f    // 1/sqrt(1+1e-5)

typedef __attribute__((ext_vector_type(16))) __bf16 bf16x16;
typedef __attribute__((ext_vector_type(8)))  float  f32x8;

// ---------------- device helpers ----------------
__device__ __forceinline__ uint16_t f2bf(float f) {
  unsigned u = __float_as_uint(f);
  unsigned r = u + 0x7fffu + ((u >> 16) & 1u);   // round-to-nearest-even
  return (uint16_t)(r >> 16);
}
__device__ __forceinline__ float lrelu(float x) { return x > 0.f ? x : 0.2f * x; }
// monotonic float<->uint encoding for atomicMax-based segment max
__device__ __forceinline__ unsigned f2ord(float f) {
  unsigned u = __float_as_uint(f);
  return (u & 0x80000000u) ? ~u : (u | 0x80000000u);
}
__device__ __forceinline__ float ord2f(unsigned x) {
  return __uint_as_float((x & 0x80000000u) ? (x & 0x7fffffffu) : ~x);
}

// ---------------- generic elementwise ----------------
__global__ void k_cvt_bf16(const float* __restrict__ in, uint16_t* __restrict__ out, int n) {
  int t = blockIdx.x * blockDim.x + threadIdx.x;
  if (t < n) out[t] = f2bf(in[t]);
}

// ---------------- WMMA GEMM:  C[M,128] = A_bf16[M,K] @ Bpacked + bias ----------------
// block = 256 threads = 8 waves; wave w owns 16-col tile w; block owns 16 rows.
// A layout (ISA 16-bit A 16x32): lane L -> row m0+(L&15), Kbase=(L<16?0:8),
//   elements 0..7 = K[Kb..Kb+7], elements 8..15 = K[Kb+16..Kb+23] (two 16B runs).
// B packed (ISA 16-bit B 32x16): lane L -> col (L&15), Kbase=(L<16?0:8),
//   elements 0..7 = K[Kb..Kb+7], elements 8..15 = K[Kb+16..Kb+23] (pre-packed by k_pack_w).
union ABu { uint4 u[2]; bf16x16 v; };

__global__ void k_wmma_gemm(const uint16_t* __restrict__ A, int K,
                            const uint16_t* __restrict__ Bp,
                            const float* __restrict__ bias,
                            float* __restrict__ Cf, uint16_t* __restrict__ Cbf) {
  const int lane = threadIdx.x & 31;
  const int wid  = threadIdx.x >> 5;
  const int m0   = blockIdx.x << 4;
  f32x8 acc = {};
  const uint16_t* Ap = A + (size_t)(m0 + (lane & 15)) * K + ((lane < 16) ? 0 : 8);
  const uint16_t* Bw = Bp + ((size_t)wid * 32 + lane) * 16;
  for (int kt = 0; kt < K; kt += 32) {
    ABu a, b;
    a.u[0] = *(const uint4*)(Ap + kt);
    a.u[1] = *(const uint4*)(Ap + kt + 16);
    b.u[0] = *(const uint4*)(Bw);
    b.u[1] = *(const uint4*)(Bw + 8);
    Bw += 8 * 32 * 16;
    __builtin_prefetch(Ap + kt + 32, 0, 0);
    acc = __builtin_amdgcn_wmma_f32_16x16x32_bf16(false, a.v, false, b.v,
                                                  (short)0, acc, false, false);
  }
  // C/D layout: lane -> col (lane&15); VGPR r -> row m0 + r + (lane>=16 ? 8 : 0)
  const int col = (wid << 4) + (lane & 15);
  const float bv = bias ? bias[col] : 0.0f;
  const int mbase = m0 + ((lane >> 4) << 3);
#pragma unroll
  for (int r = 0; r < 8; ++r) {
    float v = acc[r] + bv;
    size_t idx = (size_t)(mbase + r) * HID + col;
    Cf[idx] = v;
    if (Cbf) Cbf[idx] = f2bf(v);
  }
}

// pack row-major f32 W[K][128] into the B operand layout (bf16)
__global__ void k_pack_w(const float* __restrict__ W, uint16_t* __restrict__ Bp, int K) {
  int t = blockIdx.x * blockDim.x + threadIdx.x;
  if (t >= K * 128) return;
  int idx  = t & 15;
  int lane = (t >> 4) & 31;
  int nt   = (t >> 9) & 7;
  int kt   = t >> 12;
  int k = kt * 32 + ((lane < 16) ? 0 : 8) + idx + ((idx < 8) ? 0 : 8);
  int n = nt * 16 + (lane & 15);
  Bp[t] = f2bf(W[k * 128 + n]);
}

// ---------------- folded edge-feature attention path ----------------
// we_a[L][k][h] = sum_c We_L[k][h*32+c] * a_e_L[h][c]
__global__ void k_we_a(const float* __restrict__ We0, const float* __restrict__ We1,
                       const float* __restrict__ We2, const float* __restrict__ ae0,
                       const float* __restrict__ ae1, const float* __restrict__ ae2,
                       float* __restrict__ wea) {
  int t = blockIdx.x * blockDim.x + threadIdx.x;
  if (t >= 3 * 128 * 4) return;
  int h = t & 3, k = (t >> 2) & 127, L = t >> 9;
  const float* We = (L == 0) ? We0 : (L == 1) ? We1 : We2;
  const float* ae = (L == 0) ? ae0 : (L == 1) ? ae1 : ae2;
  float s = 0.f;
  for (int c = 0; c < 32; ++c) s += We[k * 128 + h * 32 + c] * ae[h * 32 + c];
  wea[t] = s;
}
// Mbuf[L][k<16][h] = edge_W[k]·we_a[L][:,h];  cc[L][h] = edge_b·we_a[L][:,h]
__global__ void k_M_cc(const float* __restrict__ eW, const float* __restrict__ eb,
                       const float* __restrict__ wea, float* __restrict__ Mbuf,
                       float* __restrict__ ccbuf) {
  int t = blockIdx.x * blockDim.x + threadIdx.x;
  if (t < 192) {
    int h = t & 3, k = (t >> 2) & 15, L = t >> 6;
    float s = 0.f;
    for (int j = 0; j < 128; ++j) s += eW[k * 128 + j] * wea[((L << 7) + j) * 4 + h];
    Mbuf[t] = s;
  } else if (t < 204) {
    int u = t - 192, h = u & 3, L = u >> 2;
    float s = 0.f;
    for (int j = 0; j < 128; ++j) s += eb[j] * wea[((L << 7) + j) * 4 + h];
    ccbuf[u] = s;
  }
}
// column sums of ea[E,16] (for self-loop mean row)
__global__ void k_colsum(const float* __restrict__ ea, float* __restrict__ colsum) {
  __shared__ float sh[128];
  int lane = threadIdx.x & 15, grp = threadIdx.x >> 4;
  float s = 0.f;
  for (int r = blockIdx.x * 8 + grp; r < N_EDGES; r += gridDim.x * 8)
    s += ea[(size_t)r * 16 + lane];
  sh[threadIdx.x] = s;
  __syncthreads();
  if (threadIdx.x < 16) {
    float tt = 0.f;
    for (int g = 0; g < 8; ++g) tt += sh[g * 16 + threadIdx.x];
    atomicAdd(&colsum[threadIdx.x], tt);
  }
}
// se_loop[L][h] = mean_ea·Mbuf[L][:,h] + cc[L][h]
__global__ void k_se_loop(const float* __restrict__ colsum, const float* __restrict__ Mbuf,
                          const float* __restrict__ cc, float* __restrict__ seloop) {
  int t = threadIdx.x;
  if (t >= 12) return;
  int h = t & 3, L = t >> 2;
  float s = 0.f;
  for (int k = 0; k < 16; ++k) s += (colsum[k] / (float)N_EDGES) * Mbuf[(L * 16 + k) * 4 + h];
  seloop[t] = s + cc[t];
}
// se[L][e][h] for real edges, all 3 layers in one pass over ea
__global__ void k_edge_se(const float* __restrict__ ea, const float* __restrict__ Mbuf,
                          const float* __restrict__ cc, float* __restrict__ se) {
  int e = blockIdx.x * blockDim.x + threadIdx.x;
  if (e >= N_EDGES) return;
  float v[16];
  const float4* p = (const float4*)(ea + (size_t)e * 16);
#pragma unroll
  for (int q = 0; q < 4; ++q) {
    float4 x = p[q];
    v[4 * q] = x.x; v[4 * q + 1] = x.y; v[4 * q + 2] = x.z; v[4 * q + 3] = x.w;
  }
#pragma unroll
  for (int L = 0; L < 3; ++L) {
    float4 s4;
    float* sp = &s4.x;
#pragma unroll
    for (int h = 0; h < 4; ++h) {
      float acc = cc[L * 4 + h];
#pragma unroll
      for (int k = 0; k < 16; ++k) acc += v[k] * Mbuf[(L * 16 + k) * 4 + h];
      sp[h] = acc;
    }
    *(float4*)(se + ((size_t)L * N_EDGES + e) * 4) = s4;
  }
}

// ---------------- per-layer node scores + accumulator init (wave per node) ----------------
__global__ void k_node_scores(const float* __restrict__ xl, const float* __restrict__ a_src,
                              const float* __restrict__ a_dst, float* __restrict__ ssrc,
                              float* __restrict__ sdst, unsigned* __restrict__ amax,
                              float* __restrict__ den, float* __restrict__ outacc) {
  int lane = threadIdx.x & 31, wid = threadIdx.x >> 5;
  int n = blockIdx.x * 8 + wid;
  if (n >= N_NODES) return;
  float4 x = *(const float4*)(xl + (size_t)n * HID + lane * 4);
  int h = lane >> 3, cb = (lane & 7) * 4;
  const float* as = a_src + h * 32 + cb;
  const float* ad = a_dst + h * 32 + cb;
  float ps = x.x * as[0] + x.y * as[1] + x.z * as[2] + x.w * as[3];
  float pd = x.x * ad[0] + x.y * ad[1] + x.z * ad[2] + x.w * ad[3];
  ps += __shfl_xor(ps, 1, 32); pd += __shfl_xor(pd, 1, 32);
  ps += __shfl_xor(ps, 2, 32); pd += __shfl_xor(pd, 2, 32);
  ps += __shfl_xor(ps, 4, 32); pd += __shfl_xor(pd, 4, 32);
  if ((lane & 7) == 0) {
    ssrc[n * 4 + h] = ps; sdst[n * 4 + h] = pd;
    amax[n * 4 + h] = 0u; den[n * 4 + h] = 0.f;
  }
  *(float4*)(outacc + (size_t)n * HID + lane * 4) = make_float4(0.f, 0.f, 0.f, 0.f);
}

// ---------------- attention pass A: segment max via encoded atomicMax ----------------
__global__ void k_attn_max(const int* __restrict__ src, const int* __restrict__ dst,
                           const float* __restrict__ ssrc, const float* __restrict__ sdst,
                           const float* __restrict__ se, const float* __restrict__ seloop,
                           unsigned* __restrict__ amax) {
  int e = blockIdx.x * blockDim.x + threadIdx.x;
  if (e >= N_TOT) return;
  int s, d; float4 ae;
  if (e < N_EDGES) { s = src[e]; d = dst[e]; ae = *(const float4*)(se + (size_t)e * 4); }
  else             { s = d = e - N_EDGES;    ae = *(const float4*)seloop; }
  float4 as = *(const float4*)(ssrc + (size_t)s * 4);
  float4 ad = *(const float4*)(sdst + (size_t)d * 4);
  atomicMax(&amax[d * 4 + 0], f2ord(lrelu(as.x + ad.x + ae.x)));
  atomicMax(&amax[d * 4 + 1], f2ord(lrelu(as.y + ad.y + ae.y)));
  atomicMax(&amax[d * 4 + 2], f2ord(lrelu(as.z + ad.z + ae.z)));
  atomicMax(&amax[d * 4 + 3], f2ord(lrelu(as.w + ad.w + ae.w)));
}

// ---------------- attention pass B: exp + denominator + value scatter (wave/edge) -------
__global__ void k_attn_aggr(const int* __restrict__ src, const int* __restrict__ dst,
                            const float* __restrict__ ssrc, const float* __restrict__ sdst,
                            const float* __restrict__ se, const float* __restrict__ seloop,
                            const unsigned* __restrict__ amax, const float* __restrict__ xl,
                            float* __restrict__ den, float* __restrict__ outacc) {
  int lane = threadIdx.x & 31, wid = threadIdx.x >> 5;
  int e = blockIdx.x * 8 + wid;
  if (e >= N_TOT) return;
  int s, d; float4 ae;
  if (e < N_EDGES) { s = src[e]; d = dst[e]; ae = *(const float4*)(se + (size_t)e * 4); }
  else             { s = d = e - N_EDGES;    ae = *(const float4*)seloop; }
  float4 as = *(const float4*)(ssrc + (size_t)s * 4);
  float4 ad = *(const float4*)(sdst + (size_t)d * 4);
  uint4  mx = *(const uint4*)(amax + (size_t)d * 4);
  float e0 = __expf(lrelu(as.x + ad.x + ae.x) - ord2f(mx.x));
  float e1 = __expf(lrelu(as.y + ad.y + ae.y) - ord2f(mx.y));
  float e2 = __expf(lrelu(as.z + ad.z + ae.z) - ord2f(mx.z));
  float e3 = __expf(lrelu(as.w + ad.w + ae.w) - ord2f(mx.w));
  if (lane < 4) {
    float ev = (lane == 0) ? e0 : (lane == 1) ? e1 : (lane == 2) ? e2 : e3;
    atomicAdd(&den[d * 4 + lane], ev);
  }
  const float* xs = xl + (size_t)s * HID;
  float* od = outacc + (size_t)d * HID;
  atomicAdd(&od[lane],       e0 * xs[lane]);
  atomicAdd(&od[32 + lane],  e1 * xs[32 + lane]);
  atomicAdd(&od[64 + lane],  e2 * xs[64 + lane]);
  atomicAdd(&od[96 + lane],  e3 * xs[96 + lane]);
}

// ---------------- node epilogue: normalize + bias + relu + BN -> f32 & bf16 --------------
__global__ void k_node_update(const float* __restrict__ outacc, const float* __restrict__ den,
                              const float* __restrict__ bias, const float* __restrict__ g,
                              const float* __restrict__ b, float* __restrict__ hf,
                              uint16_t* __restrict__ hbf) {
  int t = blockIdx.x * blockDim.x + threadIdx.x;
  if (t >= N_NODES * HID) return;
  int n = t >> 7, col = t & 127;
  float v = outacc[t] / den[n * 4 + (col >> 5)] + bias[col];
  v = fmaxf(v, 0.f);
  v = v * (g[col] * BN_SCALE) + b[col];
  hf[t] = v;
  hbf[t] = f2bf(v);
}

// ---------------- mean pool ----------------
__global__ void k_pool_add(const float* __restrict__ hf, const int* __restrict__ batch,
                           float* __restrict__ pooled, float* __restrict__ cnt) {
  int t = blockIdx.x * blockDim.x + threadIdx.x;
  if (t >= N_NODES * HID) return;
  int n = t >> 7, col = t & 127;
  int b = batch[n];
  atomicAdd(&pooled[(size_t)b * HID + col], hf[t]);
  if (col == 0) atomicAdd(&cnt[b], 1.0f);
}
__global__ void k_pool_div(float* __restrict__ pooled, const float* __restrict__ cnt) {
  int t = blockIdx.x * blockDim.x + threadIdx.x;
  if (t >= NBATCH * HID) return;
  pooled[t] /= fmaxf(cnt[t >> 7], 1.0f);
}

// ---------------- small dense layers: out = act(A[M,K] @ W[K,Nc] + bias) ----------------
// mode: 0 none, 1 relu, 2 relu+eval-BN(g,bb), 3 sigmoid
__global__ void k_smm(const float* __restrict__ A, const float* __restrict__ W,
                      const float* __restrict__ bias, float* __restrict__ out,
                      int M, int K, int Nc, int mode, const float* __restrict__ g,
                      const float* __restrict__ bb) {
  int t = blockIdx.x * blockDim.x + threadIdx.x;
  if (t >= M * Nc) return;
  int row = t / Nc, col = t - row * Nc;
  float s = bias ? bias[col] : 0.f;
  const float* a = A + (size_t)row * K;
  for (int k = 0; k < K; ++k) s += a[k] * W[k * Nc + col];
  if (mode == 1 || mode == 2) s = fmaxf(s, 0.f);
  if (mode == 2) s = s * (g[col] * BN_SCALE) + bb[col];
  if (mode == 3) s = 1.f / (1.f + __expf(-s));
  out[t] = s;
}

__global__ void k_gfeat(const float* __restrict__ g1, const float* __restrict__ g2,
                        float* __restrict__ gf) {
  int t = blockIdx.x * blockDim.x + threadIdx.x;
  if (t >= NBATCH * 2 * GDIM) return;
  int b = t / (2 * GDIM), k = t - b * 2 * GDIM;
  float x1, x2;
  if (k < GDIM) { x1 = g1[(size_t)b * GDIM + k]; x2 = g2[(size_t)b * GDIM + k]; gf[t] = fabsf(x1 - x2); }
  else { int kk = k - GDIM; x1 = g1[(size_t)b * GDIM + kk]; x2 = g2[(size_t)b * GDIM + kk]; gf[t] = x1 + x2; }
}
__global__ void k_concat(const float* __restrict__ e1, const float* __restrict__ e2,
                         const float* __restrict__ gg, float* __restrict__ c) {
  int t = blockIdx.x * blockDim.x + threadIdx.x;
  if (t >= NBATCH * 90) return;
  int b = t / 90, k = t - b * 90;
  c[t] = (k < 36) ? e1[b * 36 + k] : (k < 72) ? e2[b * 36 + (k - 36)] : gg[b * 18 + (k - 72)];
}

// ============================ host orchestration ============================
// d_in layout: data inputs 0..9 in setup_inputs() insertion order, then the leaves of
// the `params` pytree in jax tree_flatten order (dict keys sorted; 'W' < 'b' < 'be' < 'g').
enum {
  IN_X1 = 0, IN_EI1, IN_EA1, IN_BATCH1, IN_G1,
  IN_X2, IN_EI2, IN_EA2, IN_BATCH2, IN_G2,
  // clf: W1 W2 W3 W4 b1 b2 b3 b4 be1 be2 g1 g2
  P_CLF_W1 = 10, P_CLF_W2, P_CLF_W3, P_CLF_W4, P_CLF_B1, P_CLF_B2, P_CLF_B3, P_CLF_B4,
  P_CLF_BE1, P_CLF_BE2, P_CLF_G1, P_CLF_G2,
  // enc.bns (b,g)x3
  P_BN0_B = 22, P_BN0_G, P_BN1_B, P_BN1_G, P_BN2_B, P_BN2_G,
  // enc.convs x3: W We a_dst a_e a_src bias
  P_C0_W = 28, P_C0_WE, P_C0_ADST, P_C0_AE, P_C0_ASRC, P_C0_BIAS,
  P_C1_W = 34, P_C1_WE, P_C1_ADST, P_C1_AE, P_C1_ASRC, P_C1_BIAS,
  P_C2_W = 40, P_C2_WE, P_C2_ADST, P_C2_AE, P_C2_ASRC, P_C2_BIAS,
  P_EDGE_W = 46, P_EDGE_B, P_FC_W, P_FC_B, P_NODE_W, P_NODE_B,
  // gfc: W1 W2 b1 b2
  P_GFC_W1 = 52, P_GFC_W2, P_GFC_B1, P_GFC_B2
};

extern "C" void kernel_launch(void* const* d_in, const int* in_sizes, int n_in,
                              void* d_out, int out_size, void* d_ws, size_t ws_size,
                              hipStream_t stream) {
  (void)in_sizes; (void)n_in; (void)out_size; (void)ws_size;
  const float* F(int i); // fwd decl silence
  auto FP = [&](int i) { return (const float*)d_in[i]; };
  auto IP = [&](int i) { return (const int*)d_in[i]; };

  // ---- workspace carve-out (256B aligned) ----
  char* base = (char*)d_ws;
  size_t off = 0;
  auto alloc = [&](size_t bytes) -> char* {
    char* p = base + off;
    off = (off + bytes + 255) & ~(size_t)255;
    return p;
  };
  uint16_t* xbf    = (uint16_t*)alloc((size_t)N_NODES * NODE_DIM * 2);
  uint16_t* hbf    = (uint16_t*)alloc((size_t)N_NODES * HID * 2);
  float*    hf     = (float*)alloc((size_t)N_NODES * HID * 4);
  float*    xl     = (float*)alloc((size_t)N_NODES * HID * 4);
  float*    outacc = (float*)alloc((size_t)N_NODES * HID * 4);
  float*    ssrc   = (float*)alloc((size_t)N_NODES * 4 * 4);
  float*    sdst   = (float*)alloc((size_t)N_NODES * 4 * 4);
  unsigned* amax   = (unsigned*)alloc((size_t)N_NODES * 4 * 4);
  float*    den    = (float*)alloc((size_t)N_NODES * 4 * 4);
  float*    se     = (float*)alloc((size_t)3 * N_EDGES * 4 * 4);
  float*    seloop = (float*)alloc(12 * 4);
  float*    colsum = (float*)alloc(16 * 4);
  float*    wea    = (float*)alloc(3 * 128 * 4 * 4);
  float*    Mbuf   = (float*)alloc(3 * 16 * 4 * 4);
  float*    ccbuf  = (float*)alloc(12 * 4);
  uint16_t* nodeWp = (uint16_t*)alloc((size_t)NODE_DIM * 128 * 2);
  uint16_t* convWp = (uint16_t*)alloc((size_t)3 * 128 * 128 * 2);
  float*    pooled = (float*)alloc((size_t)NBATCH * HID * 4);
  float*    cnt    = (float*)alloc((size_t)NBATCH * 4);
  float*    emb1   = (float*)alloc((size_t)NBATCH * EMBD * 4);
  float*    emb2   = (float*)alloc((size_t)NBATCH * EMBD * 4);
  float*    gfeat  = (float*)alloc((size_t)NBATCH * 2 * GDIM * 4);
  float*    gh1    = (float*)alloc((size_t)NBATCH * 64 * 4);
  float*    gout   = (float*)alloc((size_t)NBATCH * 18 * 4);
  float*    cbuf   = (float*)alloc((size_t)NBATCH * 90 * 4);
  float*    ch1    = (float*)alloc((size_t)NBATCH * 128 * 4);
  float*    ch2    = (float*)alloc((size_t)NBATCH * 64 * 4);
  float*    ch3    = (float*)alloc((size_t)NBATCH * 32 * 4);

  const int convW[3]  = {P_C0_W, P_C1_W, P_C2_W};
  const int convAS[3] = {P_C0_ASRC, P_C1_ASRC, P_C2_ASRC};
  const int convAD[3] = {P_C0_ADST, P_C1_ADST, P_C2_ADST};
  const int convBI[3] = {P_C0_BIAS, P_C1_BIAS, P_C2_BIAS};
  const int bnB[3] = {P_BN0_B, P_BN1_B, P_BN2_B};
  const int bnG[3] = {P_BN0_G, P_BN1_G, P_BN2_G};

  // ---- once-per-launch: weight packing + folded attention projections ----
  k_pack_w<<<(NODE_DIM * 128 + 255) / 256, 256, 0, stream>>>(FP(P_NODE_W), nodeWp, NODE_DIM);
  for (int L = 0; L < 3; ++L)
    k_pack_w<<<(128 * 128 + 255) / 256, 256, 0, stream>>>(FP(convW[L]),
                                                          convWp + (size_t)L * 128 * 128, 128);
  k_we_a<<<6, 256, 0, stream>>>(FP(P_C0_WE), FP(P_C1_WE), FP(P_C2_WE),
                                FP(P_C0_AE), FP(P_C1_AE), FP(P_C2_AE), wea);
  k_M_cc<<<1, 256, 0, stream>>>(FP(P_EDGE_W), FP(P_EDGE_B), wea, Mbuf, ccbuf);

  // ---- per-graph encoder ----
  for (int gph = 0; gph < 2; ++gph) {
    const float* x   = FP(gph ? IN_X2 : IN_X1);
    const int*   ei  = IP(gph ? IN_EI2 : IN_EI1);
    const float* ea  = FP(gph ? IN_EA2 : IN_EA1);
    const int*   bat = IP(gph ? IN_BATCH2 : IN_BATCH1);
    const int* srcI = ei;
    const int* dstI = ei + N_EDGES;
    float* embg = gph ? emb2 : emb1;

    // folded edge-feature scores (all 3 layers)
    hipMemsetAsync(colsum, 0, 16 * 4, stream);
    k_colsum<<<1024, 128, 0, stream>>>(ea, colsum);
    k_se_loop<<<1, 32, 0, stream>>>(colsum, Mbuf, ccbuf, seloop);
    k_edge_se<<<(N_EDGES + 255) / 256, 256, 0, stream>>>(ea, Mbuf, ccbuf, se);

    // node encoder: h = x @ node_W + node_b   (WMMA, writes f32 + bf16)
    k_cvt_bf16<<<(N_NODES * NODE_DIM + 255) / 256, 256, 0, stream>>>(x, xbf, N_NODES * NODE_DIM);
    k_wmma_gemm<<<N_NODES / 16, 256, 0, stream>>>(xbf, NODE_DIM, nodeWp, FP(P_NODE_B), hf, hbf);

    for (int L = 0; L < 3; ++L) {
      const float* seL = se + (size_t)L * N_EDGES * 4;
      const float* slL = seloop + L * 4;
      // xl = h @ W_L  (WMMA)
      k_wmma_gemm<<<N_NODES / 16, 256, 0, stream>>>(hbf, HID, convWp + (size_t)L * 128 * 128,
                                                    nullptr, xl, nullptr);
      k_node_scores<<<(N_NODES + 7) / 8, 256, 0, stream>>>(xl, FP(convAS[L]), FP(convAD[L]),
                                                           ssrc, sdst, amax, den, outacc);
      k_attn_max<<<(N_TOT + 255) / 256, 256, 0, stream>>>(srcI, dstI, ssrc, sdst, seL, slL, amax);
      k_attn_aggr<<<(N_TOT + 7) / 8, 256, 0, stream>>>(srcI, dstI, ssrc, sdst, seL, slL,
                                                       amax, xl, den, outacc);
      k_node_update<<<(N_NODES * HID + 255) / 256, 256, 0, stream>>>(
          outacc, den, FP(convBI[L]), FP(bnG[L]), FP(bnB[L]), hf, hbf);
    }

    // mean pool + fc -> emb
    hipMemsetAsync(pooled, 0, (size_t)NBATCH * HID * 4, stream);
    hipMemsetAsync(cnt, 0, (size_t)NBATCH * 4, stream);
    k_pool_add<<<(N_NODES * HID + 255) / 256, 256, 0, stream>>>(hf, bat, pooled, cnt);
    k_pool_div<<<(NBATCH * HID + 255) / 256, 256, 0, stream>>>(pooled, cnt);
    k_smm<<<(NBATCH * EMBD + 255) / 256, 256, 0, stream>>>(pooled, FP(P_FC_W), FP(P_FC_B),
                                                           embg, NBATCH, HID, EMBD, 0,
                                                           nullptr, nullptr);
  }

  // ---- global-feature MLP + classifier ----
  k_gfeat<<<(NBATCH * 2 * GDIM + 255) / 256, 256, 0, stream>>>(FP(IN_G1), FP(IN_G2), gfeat);
  k_smm<<<(NBATCH * 64 + 255) / 256, 256, 0, stream>>>(gfeat, FP(P_GFC_W1), FP(P_GFC_B1),
                                                       gh1, NBATCH, 2 * GDIM, 64, 1, nullptr, nullptr);
  k_smm<<<(NBATCH * 18 + 255) / 256, 256, 0, stream>>>(gh1, FP(P_GFC_W2), FP(P_GFC_B2),
                                                       gout, NBATCH, 64, 18, 0, nullptr, nullptr);
  k_concat<<<(NBATCH * 90 + 255) / 256, 256, 0, stream>>>(emb1, emb2, gout, cbuf);
  k_smm<<<(NBATCH * 128 + 255) / 256, 256, 0, stream>>>(cbuf, FP(P_CLF_W1), FP(P_CLF_B1),
                                                        ch1, NBATCH, 90, 128, 2,
                                                        FP(P_CLF_G1), FP(P_CLF_BE1));
  k_smm<<<(NBATCH * 64 + 255) / 256, 256, 0, stream>>>(ch1, FP(P_CLF_W2), FP(P_CLF_B2),
                                                       ch2, NBATCH, 128, 64, 2,
                                                       FP(P_CLF_G2), FP(P_CLF_BE2));
  k_smm<<<(NBATCH * 32 + 255) / 256, 256, 0, stream>>>(ch2, FP(P_CLF_W3), FP(P_CLF_B3),
                                                       ch3, NBATCH, 64, 32, 1, nullptr, nullptr);
  k_smm<<<(NBATCH + 255) / 256, 256, 0, stream>>>(ch3, FP(P_CLF_W4), FP(P_CLF_B4),
                                                  (float*)d_out, NBATCH, 32, 1, 3,
                                                  nullptr, nullptr);
}
